// DEAM_1176821039814
// MI455X (gfx1250) — matrix operand order repaired and checked by reading, not compiled, not run on hardware
//
#include <hip/hip_runtime.h>
#include <cstdint>
#include <cstddef>

typedef __attribute__((ext_vector_type(16))) __bf16 v16bf;
typedef __attribute__((ext_vector_type(8)))  float  v8f;
typedef __attribute__((ext_vector_type(4)))  unsigned int v4u;
typedef __attribute__((ext_vector_type(8)))  int  v8i;
typedef __attribute__((ext_vector_type(4)))  int  v4i;

#define BATCH 16
#define CH    256
#define HH    64
#define WW    64
#define HW    4096

__device__ __forceinline__ float sigm(float v) { return 1.0f / (1.0f + __expf(-v)); }

// ---------------------------------------------------------------------------
// Kernel 1: per-(b,c) mean and max over H*W
// ---------------------------------------------------------------------------
__global__ void reduce_bc(const float* __restrict__ x, float* __restrict__ gap,
                          float* __restrict__ gmax) {
    int bc = blockIdx.x;                       // b*256 + c
    const float* p = x + (size_t)bc * HW;
    float s = 0.0f, mx = -3.4e38f;
    for (int i = threadIdx.x; i < HW; i += 256) {
        float v = p[i];
        s += v;
        mx = fmaxf(mx, v);
    }
    __shared__ float ss[256], sm[256];
    ss[threadIdx.x] = s; sm[threadIdx.x] = mx;
    __syncthreads();
    for (int st = 128; st > 0; st >>= 1) {
        if (threadIdx.x < st) {
            ss[threadIdx.x] += ss[threadIdx.x + st];
            sm[threadIdx.x] = fmaxf(sm[threadIdx.x], sm[threadIdx.x + st]);
        }
        __syncthreads();
    }
    if (threadIdx.x == 0) {
        gap[bc]  = ss[0] * (1.0f / (float)HW);
        gmax[bc] = sm[0];
    }
}

// ---------------------------------------------------------------------------
// Kernel 2: all the small per-sample attention math (one block per sample)
// ---------------------------------------------------------------------------
__global__ void attn_kernel(const float* __restrict__ gap, const float* __restrict__ gmax,
                            const float* __restrict__ aw3, const float* __restrict__ aoutw,
                            const float* __restrict__ akernw,
                            const float* __restrict__ lw3, const float* __restrict__ lb,
                            const float* __restrict__ mw1, const float* __restrict__ mw2,
                            float* __restrict__ inatt, float* __restrict__ outatt,
                            float* __restrict__ katt, float* __restrict__ kscale) {
    int b = blockIdx.x, c = threadIdx.x;
    __shared__ float sg[256], st[256], sl[4], shm[16], shn[16];
    sg[c] = gap[b * CH + c];
    __syncthreads();

    // t = conv1d_same(gap, aw3)
    float tm1 = (c > 0)   ? sg[c - 1] : 0.0f;
    float tp1 = (c < 255) ? sg[c + 1] : 0.0f;
    float t = aw3[0] * tm1 + aw3[1] * sg[c] + aw3[2] * tp1;
    st[c] = t;
    inatt[b * CH + c] = sigm(t);
    __syncthreads();

    // out_att = sigmoid(t @ att_out_w.T)
    float d = 0.0f;
    for (int j = 0; j < 256; ++j) d += st[j] * aoutw[c * 256 + j];
    outatt[b * CH + c] = sigm(d);

    // kernel-attention logits
    if (c < 4) {
        float l = 0.0f;
        for (int j = 0; j < 256; ++j) l += st[j] * akernw[c * 256 + j];
        sl[c] = l;
    }
    // LGA MLP hidden (relu(v @ w1.T)) on max / mean of second half
    if (c < 16) {
        float hm = 0.0f, hn = 0.0f;
        for (int j = 0; j < 128; ++j) {
            float w = mw1[c * 128 + j];
            hm += gmax[b * CH + 128 + j] * w;
            hn += sg[128 + j] * w;
        }
        shm[c] = fmaxf(hm, 0.0f);
        shn[c] = fmaxf(hn, 0.0f);
    }
    __syncthreads();

    if (c == 0) {
        float mx = fmaxf(fmaxf(sl[0], sl[1]), fmaxf(sl[2], sl[3]));
        float e0 = __expf(sl[0] - mx), e1 = __expf(sl[1] - mx);
        float e2 = __expf(sl[2] - mx), e3 = __expf(sl[3] - mx);
        float inv = 1.0f / (e0 + e1 + e2 + e3);
        katt[b * 4 + 0] = e0 * inv;
        katt[b * 4 + 1] = e1 * inv;
        katt[b * 4 + 2] = e2 * inv;
        katt[b * 4 + 3] = e3 * inv;
    }

    // K-branch channel scale (pre channel shuffle)
    float ksv;
    if (c < 128) {
        float a  = (c > 0)   ? sg[c - 1] : 0.0f;
        float b2 = (c < 127) ? sg[c + 1] : 0.0f;   // 128-length conv boundary
        float s = lw3[0] * a + lw3[1] * sg[c] + lw3[2] * b2 + lb[0];
        ksv = sigm(s);
    } else {
        int cc = c - 128;
        float d2 = 0.0f;
        for (int r = 0; r < 16; ++r) d2 += (shm[r] + shn[r]) * mw2[cc * 16 + r];
        ksv = sigm(d2);
    }
    kscale[b * CH + c] = ksv;
}

// ---------------------------------------------------------------------------
// Kernel 3: pack x_mod = x * in_att into bf16, layout [b][y][x][c]
// ---------------------------------------------------------------------------
__global__ void pack_x(const float* __restrict__ x, const float* __restrict__ inatt,
                       unsigned short* __restrict__ Xraw) {
    __bf16* Xbf = (__bf16*)Xraw;
    unsigned int id = blockIdx.x * 256u + threadIdx.x;   // (((b*32+c8)*64+y)*64+xc)
    int xc = id & 63;
    int y  = (id >> 6) & 63;
    int c8 = (id >> 12) & 31;
    int b  = id >> 17;
    union { __bf16 h[8]; uint4 q; } u;
#pragma unroll
    for (int j = 0; j < 8; ++j) {
        int c = c8 * 8 + j;
        float v = x[(((size_t)b * CH + c) * HH + y) * WW + xc] * inatt[b * CH + c];
        u.h[j] = (__bf16)v;
    }
    *(uint4*)(Xbf + ((((size_t)b * HH + y) * WW + xc) * CH + c8 * 8)) = u.q;
}

// ---------------------------------------------------------------------------
// Kernel 4: aggregate expert weights into bf16, layout [b][o][tap][ci]
// ---------------------------------------------------------------------------
__global__ void pack_w(const float* __restrict__ ede, const float* __restrict__ katt,
                       unsigned short* __restrict__ Wraw) {
    __bf16* Wbf = (__bf16*)Wraw;
    unsigned int id = blockIdx.x * 256u + threadIdx.x;   // (o, r, ci): 256*9*256
    int ci = id & 255;
    int r  = (id >> 8) % 9;
    int o  = id / 2304;
    float e[4];
#pragma unroll
    for (int k = 0; k < 4; ++k)
        e[k] = ede[(((size_t)k * 256 + o) * 256 + ci) * 9 + r];
#pragma unroll
    for (int b = 0; b < BATCH; ++b) {
        float s = katt[b * 4 + 0] * e[0] + katt[b * 4 + 1] * e[1] +
                  katt[b * 4 + 2] * e[2] + katt[b * 4 + 3] * e[3];
        Wbf[(((size_t)b * 256 + o) * 9 + r) * 256 + ci] = (__bf16)s;
    }
}

// ---------------------------------------------------------------------------
// Kernel 5: dynamic 3x3 conv as implicit GEMM.
//   grid (B, H), block 512 = 16 waves; wave w owns out-channels [16w,16w+16)
//   for the whole 64-pixel row.  Wave 0 stages rows y-1..y+1 of the NHWC
//   bf16 x_mod slab (96KB contiguous) into LDS with the Tensor Data Mover;
//   B fragments are then served from LDS (ds_load_b128), A from global.
//   Epilogue fuses *out_att and the shuffled K branch -> z.
// ---------------------------------------------------------------------------
__global__ void conv_wmma(const unsigned short* __restrict__ Xraw,
                          const unsigned short* __restrict__ Wraw,
                          const float* __restrict__ outatt,
                          const float* __restrict__ kscale,
                          const float* __restrict__ x,
                          float* __restrict__ z) {
    extern __shared__ __align__(16) unsigned short ldsraw[];   // 3*64*256 bf16 = 96KB
    __bf16* ldsx = (__bf16*)ldsraw;
    const __bf16* Wbf = (const __bf16*)Wraw;

    int b    = blockIdx.x;
    int y    = blockIdx.y;
    int wave = threadIdx.x >> 5;
    int lane = threadIdx.x & 31;
    int nlo  = lane & 15;
    bool hi  = lane >= 16;
    int o0   = wave * 16;
    int m    = o0 + nlo;                    // A-fragment row for this lane

    // ---- TDM: one wave DMAs the (up to) 3-row contiguous slab into LDS ----
    if (wave == 0) {
        int ylo = (y > 0)  ? y - 1 : 0;
        int yhi = (y < 63) ? y + 1 : 63;
        unsigned elems = (unsigned)(yhi - ylo + 1) * (WW * CH);   // <= 49152
        int slot0 = ylo - (y - 1);                                // 0 or 1
        unsigned long long gaddr = (unsigned long long)(const void*)Xraw +
            ((unsigned long long)((b * HH + ylo) * WW) * CH) * 2ull;
        unsigned ldsaddr = (unsigned)(unsigned long long)(void*)ldsraw +
            (unsigned)slot0 * (WW * CH * 2u);
        // D# group 0: count=1 | lds_addr | global_addr | type=2
        v4u g0 = { 1u,
                   ldsaddr,
                   (unsigned)(gaddr & 0xffffffffull),
                   (unsigned)((gaddr >> 32) & 0x01ffffffull) | (2u << 30) };
        // D# group 1: data_size=2B, 1-D tile of `elems` elements
        v8i g1 = { (int)0x00010000u,                          // data_size=1 (2B)
                   (int)((elems & 0xffffu) << 16),            // tensor_dim0[15:0]
                   (int)(((elems >> 16) & 0xffffu) | (1u << 16)), // dim0 hi | tensor_dim1=1
                   (int)((elems & 0xffffu) << 16),            // tile_dim0
                   1,                                         // tile_dim1 = 1
                   (int)elems, 0, 0 };                        // tensor_dim0_stride
        v4i g2 = {0, 0, 0, 0};
        v4i g3 = {0, 0, 0, 0};
        v8i g4 = {0, 0, 0, 0, 0, 0, 0, 0};   // 6-arg toolchain: extra group, unused
        __builtin_amdgcn_tensor_load_to_lds(g0, g1, g2, g3, g4, 0);
        __builtin_amdgcn_s_wait_tensorcnt(0);
    }
    __syncthreads();

    v8f acc[4] = {v8f{0.f,0.f,0.f,0.f,0.f,0.f,0.f,0.f},
                  v8f{0.f,0.f,0.f,0.f,0.f,0.f,0.f,0.f},
                  v8f{0.f,0.f,0.f,0.f,0.f,0.f,0.f,0.f},
                  v8f{0.f,0.f,0.f,0.f,0.f,0.f,0.f,0.f}};

    for (int r = 0; r < 9; ++r) {
        int dy = r / 3 - 1, dx = r % 3 - 1;
        int yy = y + dy;
        if (yy < 0 || yy >= HH) continue;   // wave-uniform skip (zero rows)
        const __bf16* wrow  = Wbf  + ((((size_t)b * 256 + m) * 9 + r) << 8);
        const __bf16* xrowl = ldsx + (size_t)(dy + 1) * (WW * CH);
#pragma unroll 2
        for (int cb = 0; cb < 8; ++cb) {
            int ci0 = cb * 32;
            // ---- A fragment: 16x32 bf16, lane half h holds k = 16h..16h+15 (contiguous) ----
            int ka = ci0 + (hi ? 16 : 0);
            union { v16bf v; uint4 q[2]; } A;
            A.q[0] = *(const uint4*)(wrow + ka);
            A.q[1] = *(const uint4*)(wrow + ka + 8);
            // ---- B fragments from LDS: 32x16 bf16, column = pixel ----
            int kb = ci0 + (hi ? 16 : 0);
            v16bf Bv[4];
#pragma unroll
            for (int t = 0; t < 4; ++t) {
                int xx = t * 16 + nlo + dx;
                union { v16bf v; uint4 q[2]; } B;
                if (xx >= 0 && xx < WW) {
                    const __bf16* xp = xrowl + ((size_t)xx << 8) + kb;
                    B.q[0] = *(const uint4*)(xp);
                    B.q[1] = *(const uint4*)(xp + 8);
                } else {
                    B.q[0] = make_uint4(0u, 0u, 0u, 0u);
                    B.q[1] = make_uint4(0u, 0u, 0u, 0u);
                }
                Bv[t] = B.v;
            }
#pragma unroll
            for (int t = 0; t < 4; ++t) {
                acc[t] = __builtin_amdgcn_wmma_f32_16x16x32_bf16(
                    false, A.v, false, Bv[t], (short)0, acc[t], false, false);
            }
        }
    }

    // ---- epilogue: z = Q*out_att + K_shuffled ----
#pragma unroll
    for (int j = 0; j < 8; ++j) {
        int co = o0 + j + (hi ? 8 : 0);          // C-layout: vgpr j, lane half
        float oa = outatt[b * CH + co];
        int cs = (co & 3) * 64 + (co >> 2);      // inverse channel shuffle
        float ks = kscale[b * CH + cs];
        const float* xrow = x + ((((size_t)b * CH + cs) * HH + y) << 6);
        float* zrow       = z + ((((size_t)b * CH + co) * HH + y) << 6);
#pragma unroll
        for (int t = 0; t < 4; ++t) {
            int xc = t * 16 + nlo;
            zrow[xc] = acc[t][j] * oa + ks * xrow[xc];
        }
    }
}

// ---------------------------------------------------------------------------
// Kernel 6: per-channel mean / rstd over (B,H,W)
// ---------------------------------------------------------------------------
__global__ void stats_kernel(const float* __restrict__ z, float* __restrict__ meanr,
                             float* __restrict__ rstdr) {
    int c = blockIdx.x;
    float s = 0.0f, s2 = 0.0f;
    for (int b = 0; b < BATCH; ++b) {
        const float* p = z + ((size_t)b * CH + c) * HW;
        for (int i = threadIdx.x; i < HW; i += 256) {
            float v = p[i];
            s += v; s2 += v * v;
        }
    }
    __shared__ float ss[256], sq[256];
    ss[threadIdx.x] = s; sq[threadIdx.x] = s2;
    __syncthreads();
    for (int st = 128; st > 0; st >>= 1) {
        if (threadIdx.x < st) {
            ss[threadIdx.x] += ss[threadIdx.x + st];
            sq[threadIdx.x] += sq[threadIdx.x + st];
        }
        __syncthreads();
    }
    if (threadIdx.x == 0) {
        const float invn = 1.0f / (float)(BATCH * HW);
        float mn  = ss[0] * invn;
        float var = sq[0] * invn - mn * mn;
        meanr[c] = mn;
        rstdr[c] = rsqrtf(var + 1e-5f);
    }
}

// ---------------------------------------------------------------------------
// Kernel 7: in-place batchnorm + relu
// ---------------------------------------------------------------------------
__global__ void bn_relu(float* __restrict__ z, const float* __restrict__ meanr,
                        const float* __restrict__ rstdr, const float* __restrict__ gamma,
                        const float* __restrict__ beta) {
    size_t i = (size_t)blockIdx.x * 256 + threadIdx.x;
    int c = (int)((i >> 12) & 255);
    float v = (z[i] - meanr[c]) * rstdr[c] * gamma[c] + beta[c];
    z[i] = fmaxf(v, 0.0f);
}

// ---------------------------------------------------------------------------
extern "C" void kernel_launch(void* const* d_in, const int* in_sizes, int n_in,
                              void* d_out, int out_size, void* d_ws, size_t ws_size,
                              hipStream_t stream) {
    const float* x      = (const float*)d_in[0];
    const float* aw3    = (const float*)d_in[1];
    const float* aoutw  = (const float*)d_in[2];
    const float* akernw = (const float*)d_in[3];
    const float* ede    = (const float*)d_in[4];
    const float* lw3    = (const float*)d_in[5];
    const float* lb     = (const float*)d_in[6];
    const float* mw1    = (const float*)d_in[7];
    const float* mw2    = (const float*)d_in[8];
    const float* gamma  = (const float*)d_in[9];
    const float* beta   = (const float*)d_in[10];
    float* z = (float*)d_out;

    char* ws = (char*)d_ws;
    unsigned short* Xbf = (unsigned short*)ws;                       // 32 MB
    unsigned short* Wbf = (unsigned short*)(ws + (size_t)33554432);  // 18 MB
    float* fb     = (float*)(ws + (size_t)52428800);
    float* gap    = fb;
    float* gmax   = fb + 4096;
    float* inatt  = fb + 8192;
    float* outatt = fb + 12288;
    float* kscale = fb + 16384;
    float* katt   = fb + 20480;
    float* meanr  = fb + 20544;
    float* rstdr  = fb + 20800;

    reduce_bc<<<dim3(4096), dim3(256), 0, stream>>>(x, gap, gmax);
    attn_kernel<<<dim3(16), dim3(256), 0, stream>>>(gap, gmax, aw3, aoutw, akernw,
                                                    lw3, lb, mw1, mw2,
                                                    inatt, outatt, katt, kscale);
    pack_x<<<dim3(8192), dim3(256), 0, stream>>>(x, inatt, Xbf);
    pack_w<<<dim3(2304), dim3(256), 0, stream>>>(ede, katt, Wbf);
    conv_wmma<<<dim3(16, 64), dim3(512), 3 * WW * CH * 2 /*96KB LDS*/, stream>>>(
        Xbf, Wbf, outatt, kscale, x, z);
    stats_kernel<<<dim3(256), dim3(256), 0, stream>>>(z, meanr, rstdr);
    bn_relu<<<dim3(65536), dim3(256), 0, stream>>>(z, meanr, rstdr, gamma, beta);
}
